// Attention_25855703122265
// MI455X (gfx1250) — matrix operand order repaired and checked by reading, not compiled, not run on hardware
//
#include <hip/hip_runtime.h>
#include <hip/hip_bf16.h>
#include <cstdint>
#include <cstddef>

// ---------------------------------------------------------------------------
// Types for CDNA5 WMMA (wave32): D(16x16 f32) = A(16x32 bf16) x B(32x16 bf16) + C
// ---------------------------------------------------------------------------
typedef __attribute__((ext_vector_type(16))) __bf16 v16bf;
typedef __attribute__((ext_vector_type(8)))  float  v8f;
typedef __attribute__((ext_vector_type(4)))  int    v4i;

#define D_MODEL 2048
#define SEQ     2048
#define NB      2
#define NH      16
#define HD      128
#define MTOT    (NB*SEQ)        /* 4096 rows */

__device__ __forceinline__ __bf16 f2bf(float f) {
  // round-to-nearest-even fp32 -> bf16, pure integer ops
  unsigned u = __builtin_bit_cast(unsigned, f);
  unsigned r = u + 0x7FFFu + ((u >> 16) & 1u);
  unsigned short s = (unsigned short)(r >> 16);
  return __builtin_bit_cast(__bf16, s);
}

// A-fragment (16x32 bf16, ISA 7.12.2): lane l holds row l&15; VGPR0-3 = K[kb..kb+7],
// VGPR4-7 = K[kb+16..kb+23], kb = 8*(l>>4).  Caller passes lane-resolved pointer.
__device__ __forceinline__ v16bf ld_frag_a(const __bf16* p) {
  union { v4i q[2]; v16bf v; } u;
  u.q[0] = *(const v4i*)(p);
  u.q[1] = *(const v4i*)(p + 16);
  return u.v;
}
// B-fragment (32x16 bf16): lane l holds column l&15; VGPR0-3 = K[kb..kb+7],
// VGPR4-7 = K[kb+16..kb+23], kb = 8*(l>>4).
__device__ __forceinline__ v16bf ld_frag_b(const __bf16* p) {
  union { v4i q[2]; v16bf v; } u;
  u.q[0] = *(const v4i*)(p);
  u.q[1] = *(const v4i*)(p + 16);
  return u.v;
}

__device__ __forceinline__ v8f wmma_bf16(v16bf a, v16bf b, v8f c) {
  return __builtin_amdgcn_wmma_f32_16x16x32_bf16(
      /*neg_a=*/false, a, /*neg_b=*/false, b,
      /*c_mod=*/(short)0, c, /*reuse_a=*/false, /*reuse_b=*/false);
}

// ---------------------------------------------------------------------------
// Shared GEMM core: 128x128 block tile, 256 threads = 8 waves in a 2(M) x 4(N)
// wave grid; each wave computes a 64x32 sub-tile = 4x2 WMMA accumulators.
// A: [M x 2048] bf16 row-major.  Bt: [N x 2048] bf16 (W transposed).
// ---------------------------------------------------------------------------
__device__ __forceinline__ void gemm_core_128x128(
    const __bf16* __restrict__ A, const __bf16* __restrict__ Bt,
    int m0, int n0, v8f acc[4][2])
{
  const int lane = threadIdx.x & 31;
  const int wave = threadIdx.x >> 5;
  const int wm   = wave >> 2;              // 0..1
  const int wn   = wave & 3;               // 0..3
  const int arow = lane & 15;
  const int akb  = (lane >> 4) << 3;       // 0 or 8
  const int bcol = lane & 15;
  const int bkr  = (lane >> 4) << 3;       // 0 or 8

  const __bf16* ap[4];
  const __bf16* bp[2];
#pragma unroll
  for (int mi = 0; mi < 4; ++mi)
    ap[mi] = A + (size_t)(m0 + wm*64 + mi*16 + arow) * D_MODEL + akb;
#pragma unroll
  for (int ni = 0; ni < 2; ++ni)
    bp[ni] = Bt + (size_t)(n0 + wn*32 + ni*16 + bcol) * D_MODEL + bkr;

  const v8f zero = {};
#pragma unroll
  for (int mi = 0; mi < 4; ++mi)
#pragma unroll
    for (int ni = 0; ni < 2; ++ni)
      acc[mi][ni] = zero;

#pragma unroll 2
  for (int k0 = 0; k0 < D_MODEL; k0 += 32) {
    v16bf af[4], bfv[2];
#pragma unroll
    for (int mi = 0; mi < 4; ++mi) af[mi]  = ld_frag_a(ap[mi] + k0);
#pragma unroll
    for (int ni = 0; ni < 2; ++ni) bfv[ni] = ld_frag_b(bp[ni] + k0);
    if (k0 + 256 < D_MODEL) {                 // one K-block lookahead
      __builtin_prefetch(ap[0] + k0 + 256, 0, 1);
      __builtin_prefetch(bp[0] + k0 + 256, 0, 1);
    }
#pragma unroll
    for (int mi = 0; mi < 4; ++mi)
#pragma unroll
      for (int ni = 0; ni < 2; ++ni)
        acc[mi][ni] = wmma_bf16(af[mi], bfv[ni], acc[mi][ni]);
  }
}

// ---------------------------------------------------------------------------
// Kernel 1: fp32 -> bf16 convert (x -> xb), 4 elems/thread
// ---------------------------------------------------------------------------
__global__ __launch_bounds__(256) void cvt_f32_bf16(
    const float* __restrict__ x, __bf16* __restrict__ y, int n4)
{
  int i = blockIdx.x * 256 + threadIdx.x;
  if (i >= n4) return;
  const float4 v = ((const float4*)x)[i];
  union { __bf16 h[4]; short4 s4; } u;
  u.h[0] = f2bf(v.x); u.h[1] = f2bf(v.y); u.h[2] = f2bf(v.z); u.h[3] = f2bf(v.w);
  ((short4*)y)[i] = u.s4;
}

// ---------------------------------------------------------------------------
// Kernel 2: transpose+convert W[k][n] fp32 -> Wt[n][k] bf16, LDS 32x33 tile
// ---------------------------------------------------------------------------
__global__ __launch_bounds__(256) void wtrans_kernel(
    const float* __restrict__ W0, const float* __restrict__ W1,
    const float* __restrict__ W2, const float* __restrict__ W3,
    __bf16* __restrict__ T0, __bf16* __restrict__ T1,
    __bf16* __restrict__ T2, __bf16* __restrict__ T3)
{
  __shared__ float tile[32][33];
  const int z = blockIdx.z;
  const float* W = (z == 0) ? W0 : (z == 1) ? W1 : (z == 2) ? W2 : W3;
  __bf16*      T = (z == 0) ? T0 : (z == 1) ? T1 : (z == 2) ? T2 : T3;
  const int tx = threadIdx.x & 31;
  const int ty = threadIdx.x >> 5;   // 0..7
  const int bn = blockIdx.x * 32;    // n (cols of W)
  const int bk = blockIdx.y * 32;    // k (rows of W)
#pragma unroll
  for (int j = 0; j < 32; j += 8)
    tile[ty + j][tx] = W[(size_t)(bk + ty + j) * D_MODEL + bn + tx];
  __syncthreads();
#pragma unroll
  for (int j = 0; j < 32; j += 8)
    T[(size_t)(bn + ty + j) * D_MODEL + bk + tx] = f2bf(tile[tx][ty + j]);
}

// ---------------------------------------------------------------------------
// Kernel 3: QKV projection.  z=0:Q  z=1:K(+present)  z=2:V(->Vt +present)
// ---------------------------------------------------------------------------
__global__ __launch_bounds__(256) void qkv_gemm_kernel(
    const __bf16* __restrict__ xb,
    const __bf16* __restrict__ Wqt, const __bf16* __restrict__ Wkt,
    const __bf16* __restrict__ Wvt,
    const float* __restrict__ bq_p, const float* __restrict__ bk_p,
    const float* __restrict__ bv_p,
    __bf16* __restrict__ Qb, __bf16* __restrict__ Kb, __bf16* __restrict__ Vt,
    float* __restrict__ present)
{
  const int z = blockIdx.z;
  const __bf16* Wt  = (z == 0) ? Wqt : (z == 1) ? Wkt : Wvt;
  const float* bias = (z == 0) ? bq_p : (z == 1) ? bk_p : bv_p;
  const int m0 = blockIdx.x * 128;
  const int n0 = blockIdx.y * 128;

  v8f acc[4][2];
  gemm_core_128x128(xb, Wt, m0, n0, acc);

  const int lane = threadIdx.x & 31;
  const int wave = threadIdx.x >> 5;
  const int wm = wave >> 2, wn = wave & 3;
  const int col = lane & 15, half = lane >> 4;

#pragma unroll
  for (int mi = 0; mi < 4; ++mi) {
#pragma unroll
    for (int ni = 0; ni < 2; ++ni) {
      const int nn = n0 + wn*32 + ni*16 + col;
      const float bias_n = bias[nn];
#pragma unroll
      for (int r = 0; r < 8; ++r) {
        const int mm = m0 + wm*64 + mi*16 + r + 8*half;   // C layout: M = r + 8*half
        const float val = acc[mi][ni][r] + bias_n;
        const int bb = mm >> 11, ss = mm & 2047;          // batch, seq
        const int hh = nn >> 7,  dd = nn & 127;           // head, head-dim
        if (z == 0) {
          Qb[(size_t)mm * D_MODEL + nn] = f2bf(val);
        } else if (z == 1) {
          Kb[(size_t)mm * D_MODEL + nn] = f2bf(val);
          present[(size_t)bb*4194304u + (size_t)hh*262144u + (size_t)ss*128u + dd] = val;
        } else {
          Vt[((size_t)(bb*NH + hh)*HD + dd) * SEQ + ss] = f2bf(val);  // head-transposed V
          present[8388608u + (size_t)bb*4194304u + (size_t)hh*262144u + (size_t)ss*128u + dd] = val;
        }
      }
    }
  }
}

// ---------------------------------------------------------------------------
// Kernel 4: flash attention (causal).  1 wave = 16 query rows, key chunks of 32.
// 8 waves/block -> 128 queries/block.  No cross-wave sync (per-wave LDS region).
// ---------------------------------------------------------------------------
__global__ __launch_bounds__(256) void flash_attn_kernel(
    const __bf16* __restrict__ Qb, const __bf16* __restrict__ Kb,
    const __bf16* __restrict__ Vt, __bf16* __restrict__ attn)
{
  __shared__ __attribute__((aligned(16))) __bf16 plds[8][16 * 32];

  const int lane = threadIdx.x & 31;
  const int wave = threadIdx.x >> 5;
  const int bh = blockIdx.y;
  const int b  = bh >> 4, h = bh & 15;
  const int q0 = blockIdx.x * 128 + wave * 16;

  const int arow = lane & 15, akb = (lane >> 4) << 3;
  const int bcol = lane & 15, bkr = (lane >> 4) << 3;
  const int col  = lane & 15, half = lane >> 4;

  const __bf16* Qhead = Qb + (size_t)(b*SEQ + q0) * D_MODEL + h*HD;
  const __bf16* Khead = Kb + (size_t)(b*SEQ)      * D_MODEL + h*HD;
  const __bf16* Vhead = Vt + (size_t)(b*NH + h) * HD * SEQ;
  __bf16* pl = &plds[wave][0];

  // Q fragments resident for the whole key loop: 4 K-steps x 16x32 bf16
  v16bf qf[4];
#pragma unroll
  for (int kk = 0; kk < 4; ++kk)
    qf[kk] = ld_frag_a(Qhead + (size_t)arow * D_MODEL + kk*32 + akb);

  const v8f zero = {};
  v8f o[8];
#pragma unroll
  for (int t = 0; t < 8; ++t) o[t] = zero;
  float mrow[8], lrow[8];
#pragma unroll
  for (int r = 0; r < 8; ++r) { mrow[r] = -3.0e38f; lrow[r] = 0.f; }

  const float scale = 0.08838834764831845f;   // 1/sqrt(128)

  for (int key0 = 0; key0 < q0 + 16; key0 += 32) {
    // ---- S = Q K^T for two 16x16 key tiles ----
    v8f s0 = zero, s1 = zero;
#pragma unroll
    for (int kk = 0; kk < 4; ++kk) {
      v16bf kf0 = ld_frag_b(Khead + (size_t)(key0 + bcol)      * D_MODEL + kk*32 + bkr);
      v16bf kf1 = ld_frag_b(Khead + (size_t)(key0 + 16 + bcol) * D_MODEL + kk*32 + bkr);
      s0 = wmma_bf16(qf[kk], kf0, s0);
      s1 = wmma_bf16(qf[kk], kf1, s1);
    }

    // ---- online softmax (row stats per C-layout: row = r + 8*half across 16 lanes)
    const bool needmask = (key0 + 31 > q0);
    float alpha[8];
#pragma unroll
    for (int r = 0; r < 8; ++r) {
      float a0 = s0[r] * scale, a1 = s1[r] * scale;
      if (needmask) {
        const int q = q0 + r + 8*half;
        if (key0 + col > q)      a0 = -1e30f;
        if (key0 + 16 + col > q) a1 = -1e30f;
      }
      float v = fmaxf(a0, a1);
      v = fmaxf(v, __shfl_xor(v, 1, 32));
      v = fmaxf(v, __shfl_xor(v, 2, 32));
      v = fmaxf(v, __shfl_xor(v, 4, 32));
      v = fmaxf(v, __shfl_xor(v, 8, 32));
      const float mnew = fmaxf(mrow[r], v);
      const float al   = __expf(mrow[r] - mnew);
      const float p0   = __expf(a0 - mnew);
      const float p1   = __expf(a1 - mnew);
      float su = p0 + p1;
      su += __shfl_xor(su, 1, 32);
      su += __shfl_xor(su, 2, 32);
      su += __shfl_xor(su, 4, 32);
      su += __shfl_xor(su, 8, 32);
      mrow[r]  = mnew;
      lrow[r]  = lrow[r] * al + su;
      alpha[r] = al;
      // C-layout -> row-major 16x32 bf16 tile in this wave's LDS region
      const int rr = r + 8*half;
      pl[rr*32 + col]      = f2bf(p0);
      pl[rr*32 + 16 + col] = f2bf(p1);
    }
#pragma unroll
    for (int t = 0; t < 8; ++t)
#pragma unroll
      for (int r = 0; r < 8; ++r) o[t][r] *= alpha[r];

    // intra-wave LDS RAW: DS pipe is in-order per wave; wait for counters anyway
    asm volatile("s_wait_dscnt 0\ns_wait_storecnt 0" ::: "memory");

    // ---- O += P V  (A = P 16x32 from LDS; B = V 32x16 from head-transposed Vt)
    v16bf pf = ld_frag_a(pl + arow*32 + akb);
#pragma unroll
    for (int t = 0; t < 8; ++t) {
      v16bf vf = ld_frag_b(Vhead + (size_t)(t*16 + bcol) * SEQ + key0 + bkr);
      o[t] = wmma_bf16(pf, vf, o[t]);
    }
  }

  // ---- epilogue: O /= l, merge heads, store bf16 for the output projection
#pragma unroll
  for (int t = 0; t < 8; ++t) {
#pragma unroll
    for (int r = 0; r < 8; ++r) {
      const int mm = b*SEQ + q0 + r + 8*half;
      const int nn = h*HD + t*16 + col;
      attn[(size_t)mm * D_MODEL + nn] = f2bf(o[t][r] / lrow[r]);
    }
  }
}

// ---------------------------------------------------------------------------
// Kernel 5: output projection -> fp32 'a'
// ---------------------------------------------------------------------------
__global__ __launch_bounds__(256) void out_proj_kernel(
    const __bf16* __restrict__ attn, const __bf16* __restrict__ Wdt,
    const float* __restrict__ bd_p, float* __restrict__ out)
{
  const int m0 = blockIdx.x * 128;
  const int n0 = blockIdx.y * 128;
  v8f acc[4][2];
  gemm_core_128x128(attn, Wdt, m0, n0, acc);

  const int lane = threadIdx.x & 31;
  const int wave = threadIdx.x >> 5;
  const int wm = wave >> 2, wn = wave & 3;
  const int col = lane & 15, half = lane >> 4;

#pragma unroll
  for (int mi = 0; mi < 4; ++mi) {
#pragma unroll
    for (int ni = 0; ni < 2; ++ni) {
      const int nn = n0 + wn*32 + ni*16 + col;
      const float bias_n = bd_p[nn];
#pragma unroll
      for (int r = 0; r < 8; ++r) {
        const int mm = m0 + wm*64 + mi*16 + r + 8*half;
        out[(size_t)mm * D_MODEL + nn] = acc[mi][ni][r] + bias_n;
      }
    }
  }
}

// ---------------------------------------------------------------------------
// Host launcher
// ---------------------------------------------------------------------------
extern "C" void kernel_launch(void* const* d_in, const int* in_sizes, int n_in,
                              void* d_out, int out_size, void* d_ws, size_t ws_size,
                              hipStream_t stream) {
  (void)in_sizes; (void)n_in; (void)out_size; (void)ws_size;
  const float* x  = (const float*)d_in[0];
  const float* Wq = (const float*)d_in[1];
  const float* bq = (const float*)d_in[2];
  const float* Wk = (const float*)d_in[3];
  const float* bk = (const float*)d_in[4];
  const float* Wv = (const float*)d_in[5];
  const float* bv = (const float*)d_in[6];
  const float* Wd = (const float*)d_in[7];
  const float* bd = (const float*)d_in[8];

  float* out     = (float*)d_out;
  float* present = out + (size_t)MTOT * D_MODEL;   // [2,B,H,S,hd] fp32

  char* ws = (char*)d_ws;                 // 112 MB used
  __bf16* xb  = (__bf16*)(ws);                       // 16 MB  x  bf16
  __bf16* Wqt = (__bf16*)(ws + (size_t)(16u << 20)); //  8 MB  Wq^T bf16
  __bf16* Wkt = (__bf16*)(ws + (size_t)(24u << 20));
  __bf16* Wvt = (__bf16*)(ws + (size_t)(32u << 20));
  __bf16* Wdt = (__bf16*)(ws + (size_t)(40u << 20));
  __bf16* Qb  = (__bf16*)(ws + (size_t)(48u << 20)); // 16 MB
  __bf16* Kb  = (__bf16*)(ws + (size_t)(64u << 20)); // 16 MB
  __bf16* Vt  = (__bf16*)(ws + (size_t)(80u << 20)); // 16 MB (head-transposed)
  __bf16* at  = (__bf16*)(ws + (size_t)(96u << 20)); // 16 MB merged-head attn out

  const int n4 = MTOT * D_MODEL / 4;   // 2,097,152
  cvt_f32_bf16<<<dim3((n4 + 255) / 256), dim3(256), 0, stream>>>(x, xb, n4);

  wtrans_kernel<<<dim3(64, 64, 4), dim3(256), 0, stream>>>(
      Wq, Wk, Wv, Wd, Wqt, Wkt, Wvt, Wdt);

  qkv_gemm_kernel<<<dim3(32, 16, 3), dim3(256), 0, stream>>>(
      xb, Wqt, Wkt, Wvt, bq, bk, bv, Qb, Kb, Vt, present);

  flash_attn_kernel<<<dim3(16, 32), dim3(256), 0, stream>>>(Qb, Kb, Vt, at);

  out_proj_kernel<<<dim3(32, 16), dim3(256), 0, stream>>>(at, Wdt, bd, out);
}
